// PairTransformLayer_7756710937192
// MI455X (gfx1250) — matrix operand order repaired and checked by reading, compiled, not run on hardware
//
#include <hip/hip_runtime.h>

// ---------------------------------------------------------------------------
// CDNA5 / gfx1250 PairTransformLayer.
// GEMMs: v_wmma_f32_16x16x32_bf16; weight staging: global_load_async_to_lds.
// ---------------------------------------------------------------------------

typedef __bf16 bf16_t;
typedef __attribute__((ext_vector_type(16))) __bf16 v16bf;
typedef __attribute__((ext_vector_type(8)))  __bf16 v8bf;
typedef __attribute__((ext_vector_type(8)))  float  v8f;
typedef __attribute__((ext_vector_type(4)))  int    v4i;

#define ACT_NONE 0
#define ACT_SIG  1
#define ACT_RELU 2

#if defined(__gfx1250__) && __has_builtin(__builtin_amdgcn_global_load_async_to_lds_b128) && \
    __has_builtin(__builtin_amdgcn_s_wait_asynccnt)
#define PT_HAVE_ASYNC_LDS 1
typedef __attribute__((address_space(3))) v4i as3_v4i;
#endif

static __device__ __forceinline__ float sigmoidf_(float x) {
  return __builtin_amdgcn_rcpf(1.f + __expf(-x));
}

static __device__ __forceinline__ v16bf cat8(v8bf lo, v8bf hi) {
  return __builtin_shufflevector(lo, hi, 0, 1, 2, 3, 4, 5, 6, 7,
                                         8, 9, 10, 11, 12, 13, 14, 15);
}

// ---------------------------------------------------------------------------
// Weight pre-pack: W[K][NC] f32 -> slab-swizzled bf16 fragment layout:
// out[((s*NC + n)*32) + kk] = W[(s*32+kk)*NC + n]   (s = k-slab of 32)
// Each slab is then a contiguous 64*NC-byte block, DMA-able to LDS.
// ---------------------------------------------------------------------------
__global__ void __launch_bounds__(256) pt_pack_weight(
    const float* __restrict__ W, bf16_t* __restrict__ out, int K, int NC)
{
  int nsh = 31 - __clz(NC);
  int total = K * NC;
  for (int o = blockIdx.x * 256 + threadIdx.x; o < total; o += gridDim.x * 256) {
    int kk = o & 31;
    int n  = (o >> 5) & (NC - 1);
    int s  = o >> (5 + nsh);
    out[o] = (bf16_t)W[(size_t)(s * 32 + kk) * NC + n];
  }
}

// ---------------------------------------------------------------------------
// LayerNorm over last dim (128). One wave per row, 4 channels per lane.
// ---------------------------------------------------------------------------
__global__ void __launch_bounds__(256) pt_layernorm(
    const float* __restrict__ src, float* __restrict__ dst,
    const float* __restrict__ g, const float* __restrict__ b, int rows)
{
  int row  = blockIdx.x * 8 + (threadIdx.x >> 5);
  if (row >= rows) return;
  int lane = threadIdx.x & 31;
  float4 x = *(const float4*)(src + (size_t)row * 128 + lane * 4);
  float s = x.x + x.y + x.z + x.w;
#pragma unroll
  for (int o = 1; o < 32; o <<= 1) s += __shfl_xor(s, o, 32);
  float mean = s * (1.f / 128.f);
  float dx = x.x - mean, dy = x.y - mean, dz = x.z - mean, dw = x.w - mean;
  float v = dx * dx + dy * dy + dz * dz + dw * dw;
#pragma unroll
  for (int o = 1; o < 32; o <<= 1) v += __shfl_xor(v, o, 32);
  float rstd = rsqrtf(v * (1.f / 128.f) + 1e-5f);
  float4 gg = *(const float4*)(g + lane * 4);
  float4 bb = *(const float4*)(b + lane * 4);
  float4 out;
  out.x = dx * rstd * gg.x + bb.x;
  out.y = dy * rstd * gg.y + bb.y;
  out.z = dz * rstd * gg.z + bb.z;
  out.w = dw * rstd * gg.w + bb.w;
  *(float4*)(dst + (size_t)row * 128 + lane * 4) = out;
}

// ---------------------------------------------------------------------------
// Templated fused GEMM: out = epi(A[M,K] @ W[K,NC]).
// 32 output rows per block (2 M-tiles per wave), 8 waves walk the N tiles.
// W arrives pre-packed bf16 (slab-swizzled); slabs staged to LDS by async DMA.
// ---------------------------------------------------------------------------
template <int K, int NC, int ACT, bool BIAS, bool MUL, bool MASK, bool ACCUM>
__global__ void __launch_bounds__(256) pt_gemm(
    const float* __restrict__ A, const bf16_t* __restrict__ Wswz,
    const float* __restrict__ bias, const float* __restrict__ mul,
    const float* __restrict__ msk, float* __restrict__ out, int M)
{
  constexpr int KP  = K + 8;            // padded A row (bf16 units)
  constexpr int WP  = 40;               // padded W row (bf16 units), 80B stride
  constexpr int KSH = (K == 512) ? 9 : 7;
  constexpr int TPW = NC / 128;         // N tiles per wave

  __shared__ alignas(16) bf16_t As[32 * KP];
  __shared__ alignas(16) bf16_t Ws[NC * WP];

  int row0 = blockIdx.x * 32;
  for (int idx = threadIdx.x; idx < 32 * K; idx += 256) {
    int r = idx >> KSH, c = idx & (K - 1);
    As[r * KP + c] = (bf16_t)A[(size_t)(row0 + r) * K + c];
  }

  int wave = threadIdx.x >> 5, lane = threadIdx.x & 31;
  int nlo = lane & 15, hf = lane >> 4;

  v8f acc[2][TPW];
#pragma unroll
  for (int mt = 0; mt < 2; ++mt)
#pragma unroll
    for (int t = 0; t < TPW; ++t)
      acc[mt][t] = v8f{0.f, 0.f, 0.f, 0.f, 0.f, 0.f, 0.f, 0.f};

  for (int k0 = 0; k0 < K; k0 += 32) {
    __syncthreads();                    // As ready (1st iter) / Ws reuse safe
    const bf16_t* wsrc = Wswz + (size_t)(k0 >> 5) * (NC * 32);
#ifdef PT_HAVE_ASYNC_LDS
    // DMA one 32xNC bf16 slab into the padded LDS layout: 16B chunks,
    // per-lane LDS destinations, tracked on ASYNCcnt.
    for (int idx = threadIdx.x; idx < NC * 4; idx += 256) {
      int n = idx >> 2, part = idx & 3;
      __builtin_amdgcn_global_load_async_to_lds_b128(
          (v4i*)(wsrc + (size_t)n * 32 + part * 8),
          (as3_v4i*)(&Ws[n * WP + part * 8]), 0, 0);
    }
    __builtin_amdgcn_s_wait_asynccnt(0);
#else
    for (int idx = threadIdx.x; idx < NC * 32; idx += 256) {
      int n = idx >> 5, kk = idx & 31;
      Ws[n * WP + kk] = wsrc[n * 32 + kk];
    }
#endif
    __syncthreads();

#pragma unroll
    for (int mt = 0; mt < 2; ++mt) {
      const bf16_t* ar = &As[(mt * 16 + nlo) * KP + k0 + hf * 8];
      v16bf af = cat8(*(const v8bf*)ar, *(const v8bf*)(ar + 16));
#pragma unroll
      for (int t = 0; t < TPW; ++t) {
        int n0 = (wave + t * 8) * 16;
        const bf16_t* wr = &Ws[(n0 + nlo) * WP + hf * 8];
        v16bf bfv = cat8(*(const v8bf*)wr, *(const v8bf*)(wr + 16));
        acc[mt][t] = __builtin_amdgcn_wmma_f32_16x16x32_bf16(
            false, af, false, bfv, (short)0, acc[mt][t], false, false);
      }
    }
  }

#pragma unroll
  for (int mt = 0; mt < 2; ++mt) {
#pragma unroll
    for (int t = 0; t < TPW; ++t) {
      int n0 = (wave + t * 8) * 16;
#pragma unroll
      for (int r = 0; r < 8; ++r) {
        int m = row0 + mt * 16 + hf * 8 + r;
        int n = n0 + nlo;
        size_t oi = (size_t)m * NC + n;
        float v = acc[mt][t][r];
        if constexpr (BIAS) v += bias[n];
        if constexpr (ACT == ACT_SIG)  v = sigmoidf_(v);
        if constexpr (ACT == ACT_RELU) v = fmaxf(v, 0.f);
        if constexpr (MUL)  v *= mul[oi];
        if constexpr (MASK) v *= msk[m];
        if constexpr (ACCUM) out[oi] += v; else out[oi] = v;
      }
    }
  }
  (void)M;
}

// ---------------------------------------------------------------------------
// Tiled channel-last -> channel-first transpose (both sides coalesced).
// mode 0 (outgoing): dst[c][p][q] = src[(p*256+q)*128 + c]
// mode 1 (incoming): dst[c][p][q] = src[(q*256+p)*128 + c]
// ---------------------------------------------------------------------------
__global__ void __launch_bounds__(256) pt_transpose_cf(
    const float* __restrict__ s, float* __restrict__ d, int mode)
{
  __shared__ float tile[32][33];
  int pfix = blockIdx.x;
  int q0 = blockIdx.y * 32;
  int c0 = blockIdx.z * 32;
  int tcc = threadIdx.x & 31;
  int tq  = threadIdx.x >> 5;
#pragma unroll
  for (int pass = 0; pass < 4; ++pass) {
    int q = q0 + tq + pass * 8;
    size_t pos = (mode == 0) ? ((size_t)pfix * 256 + q) : ((size_t)q * 256 + pfix);
    tile[tq + pass * 8][tcc] = s[pos * 128 + c0 + tcc];
  }
  __syncthreads();
  int tqq = threadIdx.x & 31;
  int tc  = threadIdx.x >> 5;
#pragma unroll
  for (int pass = 0; pass < 4; ++pass) {
    int c = c0 + tc + pass * 8;
    d[(size_t)c * 65536 + (size_t)pfix * 256 + q0 + tqq] = tile[tqq][tc + pass * 8];
  }
}

// ---------------------------------------------------------------------------
// Tiled channel-first -> channel-last transpose (both sides coalesced).
// ---------------------------------------------------------------------------
__global__ void __launch_bounds__(256) pt_transpose_fc(
    const float* __restrict__ s, float* __restrict__ d)
{
  __shared__ float tile[32][33];
  int ifix = blockIdx.x;
  int j0 = blockIdx.y * 32;
  int c0 = blockIdx.z * 32;
  int tj = threadIdx.x & 31;
  int tc = threadIdx.x >> 5;
#pragma unroll
  for (int pass = 0; pass < 4; ++pass) {
    int c = c0 + tc + pass * 8;
    tile[tc + pass * 8][tj] = s[(size_t)c * 65536 + (size_t)ifix * 256 + j0 + tj];
  }
  __syncthreads();
  int tcc = threadIdx.x & 31;
  int tjj = threadIdx.x >> 5;
#pragma unroll
  for (int pass = 0; pass < 4; ++pass) {
    int j = j0 + tjj + pass * 8;
    d[((size_t)ifix * 256 + j) * 128 + c0 + tcc] = tile[tcc][tjj + pass * 8];
  }
}

// ---------------------------------------------------------------------------
// Triangle einsum: per channel c, X_c[i][j] = sum_k At_c[i][k] * Bt_c[j][k].
// Writes channel-first (coalesced); a cf->cl transpose follows.
// ---------------------------------------------------------------------------
__global__ void __launch_bounds__(256) pt_tri_gemm(
    const float* __restrict__ at, const float* __restrict__ bt,
    float* __restrict__ xcf)
{
  constexpr int AP = 264;
  __shared__ alignas(16) bf16_t As2[16 * AP];

  int c = blockIdx.x;
  const float* A = at + (size_t)c * 65536;
  const float* B = bt + (size_t)c * 65536;
  int i0 = blockIdx.y * 16;

  for (int idx = threadIdx.x; idx < 16 * 256; idx += 256) {
    int r = idx >> 8, cc = idx & 255;
    As2[r * AP + cc] = (bf16_t)A[(size_t)(i0 + r) * 256 + cc];
  }
  __syncthreads();

  int wave = threadIdx.x >> 5, lane = threadIdx.x & 31;
  int nlo = lane & 15, hf = lane >> 4;
  int j0 = (blockIdx.z * 8 + wave) * 16;

  v8f acc = {0.f, 0.f, 0.f, 0.f, 0.f, 0.f, 0.f, 0.f};
  const float* Brow = B + (size_t)(j0 + nlo) * 256;
  for (int k0 = 0; k0 < 256; k0 += 32) {
    const bf16_t* ar = &As2[nlo * AP + k0 + hf * 8];
    v16bf af = cat8(*(const v8bf*)ar, *(const v8bf*)(ar + 16));
    v16bf bfv;
#pragma unroll
    for (int e = 0; e < 16; ++e) bfv[e] = (bf16_t)Brow[k0 + hf * 8 + e + (e & 8)];
    acc = __builtin_amdgcn_wmma_f32_16x16x32_bf16(
        false, af, false, bfv, (short)0, acc, false, false);
  }
#pragma unroll
  for (int r = 0; r < 8; ++r) {
    int i = i0 + hf * 8 + r;
    xcf[(size_t)c * 65536 + (size_t)i * 256 + j0 + nlo] = acc[r];
  }
}

// ---------------------------------------------------------------------------
// Triangle bias: tb2[h][pos] = sum_c zln[pos*128+c] * tbw[c*4+h]
// ---------------------------------------------------------------------------
__global__ void __launch_bounds__(256) pt_tribias(
    const float* __restrict__ zln, const float* __restrict__ tbw,
    float* __restrict__ tb2)
{
  int pos = blockIdx.x * 8 + (threadIdx.x >> 5);
  int lane = threadIdx.x & 31;
  float4 x = *(const float4*)(zln + (size_t)pos * 128 + lane * 4);
  const float* w = tbw + lane * 16;
  float a0 = x.x * w[0] + x.y * w[4] + x.z * w[8]  + x.w * w[12];
  float a1 = x.x * w[1] + x.y * w[5] + x.z * w[9]  + x.w * w[13];
  float a2 = x.x * w[2] + x.y * w[6] + x.z * w[10] + x.w * w[14];
  float a3 = x.x * w[3] + x.y * w[7] + x.z * w[11] + x.w * w[15];
#pragma unroll
  for (int o = 1; o < 32; o <<= 1) {
    a0 += __shfl_xor(a0, o, 32); a1 += __shfl_xor(a1, o, 32);
    a2 += __shfl_xor(a2, o, 32); a3 += __shfl_xor(a3, o, 32);
  }
  if (lane == 0) {
    tb2[pos]             = a0;
    tb2[65536 + pos]     = a1;
    tb2[2 * 65536 + pos] = a2;
    tb2[3 * 65536 + pos] = a3;
  }
}

// ---------------------------------------------------------------------------
// Flash attention for one (row i, head h). 8 waves x two 16-query tiles,
// online softmax; K row-major & V channel-major in LDS; P via per-wave LDS.
// ---------------------------------------------------------------------------
__global__ void __launch_bounds__(256) pt_attention(
    const float* __restrict__ qbuf, const float* __restrict__ kbuf,
    const float* __restrict__ vbuf, const float* __restrict__ tb2,
    const float* __restrict__ msk,  float* __restrict__ ob)
{
  constexpr int KSTR = 40;
  constexpr int VSTR = 264;
  constexpr int PSTR = 40;
  __shared__ alignas(16) bf16_t Ks[256 * KSTR];
  __shared__ alignas(16) bf16_t Vst[32 * VSTR];
  __shared__ alignas(16) bf16_t Ps[8 * 16 * PSTR];

  int i = blockIdx.x, h = blockIdx.y;
  const float* kp = kbuf + ((size_t)i * 256) * 128 + h * 32;
  const float* vp = vbuf + ((size_t)i * 256) * 128 + h * 32;
  for (int t = threadIdx.x; t < 256 * 32; t += 256) {
    int r = t >> 5, c = t & 31;
    Ks[r * KSTR + c] = (bf16_t)kp[(size_t)r * 128 + c];
    Vst[c * VSTR + r] = (bf16_t)vp[(size_t)r * 128 + c];
  }
  __syncthreads();

  int wave = threadIdx.x >> 5, lane = threadIdx.x & 31;
  int nlo = lane & 15, hf = lane >> 4;
  bf16_t* Pw = Ps + wave * 16 * PSTR;
  const float scale = 0.17677669529663687f; // 1/sqrt(32)
  const v8f zc = {0.f, 0.f, 0.f, 0.f, 0.f, 0.f, 0.f, 0.f};

  for (int qt = 0; qt < 2; ++qt) {
    int q0 = (wave + 8 * qt) * 16;
    const float* qrow = qbuf + ((size_t)(i * 256 + q0 + nlo)) * 128 + h * 32;
    v16bf qf;
    {
      float4 a0 = *(const float4*)(qrow + hf * 8);
      float4 a1 = *(const float4*)(qrow + hf * 8 + 4);
      float4 b0 = *(const float4*)(qrow + 16 + hf * 8);
      float4 b1 = *(const float4*)(qrow + 16 + hf * 8 + 4);
      qf[0] = (bf16_t)(a0.x * scale); qf[1] = (bf16_t)(a0.y * scale);
      qf[2] = (bf16_t)(a0.z * scale); qf[3] = (bf16_t)(a0.w * scale);
      qf[4] = (bf16_t)(a1.x * scale); qf[5] = (bf16_t)(a1.y * scale);
      qf[6] = (bf16_t)(a1.z * scale); qf[7] = (bf16_t)(a1.w * scale);
      qf[8]  = (bf16_t)(b0.x * scale); qf[9]  = (bf16_t)(b0.y * scale);
      qf[10] = (bf16_t)(b0.z * scale); qf[11] = (bf16_t)(b0.w * scale);
      qf[12] = (bf16_t)(b1.x * scale); qf[13] = (bf16_t)(b1.y * scale);
      qf[14] = (bf16_t)(b1.z * scale); qf[15] = (bf16_t)(b1.w * scale);
    }

    float mr[8], lr[8];
    v8f o0 = zc, o1 = zc;
#pragma unroll
    for (int r = 0; r < 8; ++r) { mr[r] = -1e30f; lr[r] = 0.f; }

    for (int jc = 0; jc < 8; ++jc) {
      const bf16_t* kr0 = &Ks[(jc * 32 + nlo) * KSTR + hf * 8];
      const bf16_t* kr1 = &Ks[(jc * 32 + 16 + nlo) * KSTR + hf * 8];
      v16bf kf0 = cat8(*(const v8bf*)kr0, *(const v8bf*)(kr0 + 16));
      v16bf kf1 = cat8(*(const v8bf*)kr1, *(const v8bf*)(kr1 + 16));
      v8f s0 = __builtin_amdgcn_wmma_f32_16x16x32_bf16(
          false, qf, false, kf0, (short)0, zc, false, false);
      v8f s1 = __builtin_amdgcn_wmma_f32_16x16x32_bf16(
          false, qf, false, kf1, (short)0, zc, false, false);

      float cm[8];
#pragma unroll
      for (int r = 0; r < 8; ++r) {
        int q   = q0 + hf * 8 + r;
        int k0c = jc * 32 + nlo;
        int k1c = k0c + 16;
        s0[r] += 1e9f * (msk[i * 256 + k0c] - 1.f) +
                 tb2[(size_t)h * 65536 + (size_t)q * 256 + k0c];
        s1[r] += 1e9f * (msk[i * 256 + k1c] - 1.f) +
                 tb2[(size_t)h * 65536 + (size_t)q * 256 + k1c];
        cm[r] = fmaxf(s0[r], s1[r]);
      }
#pragma unroll
      for (int r = 0; r < 8; ++r) {
#pragma unroll
        for (int o = 1; o < 16; o <<= 1) cm[r] = fmaxf(cm[r], __shfl_xor(cm[r], o, 32));
      }
      float cs[8];
#pragma unroll
      for (int r = 0; r < 8; ++r) {
        float mnew  = fmaxf(mr[r], cm[r]);
        float alpha = __expf(mr[r] - mnew);
        mr[r] = mnew;
        lr[r] *= alpha; o0[r] *= alpha; o1[r] *= alpha;
        float p0 = __expf(s0[r] - mnew);
        float p1 = __expf(s1[r] - mnew);
        s0[r] = p0; s1[r] = p1;
        cs[r] = p0 + p1;
      }
#pragma unroll
      for (int r = 0; r < 8; ++r) {
#pragma unroll
        for (int o = 1; o < 16; o <<= 1) cs[r] += __shfl_xor(cs[r], o, 32);
        lr[r] += cs[r];
      }
#pragma unroll
      for (int r = 0; r < 8; ++r) {
        Pw[(hf * 8 + r) * PSTR + nlo]      = (bf16_t)s0[r];
        Pw[(hf * 8 + r) * PSTR + 16 + nlo] = (bf16_t)s1[r];
      }
      __syncthreads();
      const bf16_t* pr = &Pw[nlo * PSTR + hf * 8];
      v16bf pf  = cat8(*(const v8bf*)pr, *(const v8bf*)(pr + 16));
      const bf16_t* vr0 = &Vst[nlo * VSTR + jc * 32 + hf * 8];
      const bf16_t* vr1 = &Vst[(16 + nlo) * VSTR + jc * 32 + hf * 8];
      v16bf vf0 = cat8(*(const v8bf*)vr0, *(const v8bf*)(vr0 + 16));
      v16bf vf1 = cat8(*(const v8bf*)vr1, *(const v8bf*)(vr1 + 16));
      o0 = __builtin_amdgcn_wmma_f32_16x16x32_bf16(
          false, pf, false, vf0, (short)0, o0, false, false);
      o1 = __builtin_amdgcn_wmma_f32_16x16x32_bf16(
          false, pf, false, vf1, (short)0, o1, false, false);
      __syncthreads();
    }

    float* op = ob + ((size_t)(i * 256 + q0)) * 128 + h * 32;
#pragma unroll
    for (int r = 0; r < 8; ++r) {
      float inv = __builtin_amdgcn_rcpf(lr[r]);
      op[(hf * 8 + r) * 128 + nlo]      = o0[r] * inv;
      op[(hf * 8 + r) * 128 + 16 + nlo] = o1[r] * inv;
    }
  }
}

// ---------------------------------------------------------------------------
// Small elementwise kernels
// ---------------------------------------------------------------------------
__global__ void __launch_bounds__(256) pt_copy(const float* __restrict__ s,
                                               float* __restrict__ d, size_t n) {
  size_t i = (size_t)blockIdx.x * 256 + threadIdx.x;
  if (i < n) d[i] = s[i];
}
__global__ void __launch_bounds__(256) pt_mul(float* __restrict__ d,
                                              const float* __restrict__ m, size_t n) {
  size_t i = (size_t)blockIdx.x * 256 + threadIdx.x;
  if (i < n) d[i] *= m[i];
}
__global__ void __launch_bounds__(256) pt_mask_rows(float* __restrict__ d,
                                                    const float* __restrict__ m, size_t n) {
  size_t i = (size_t)blockIdx.x * 256 + threadIdx.x;
  if (i < n) d[i] *= m[i >> 7];
}
__global__ void __launch_bounds__(256) pt_transpose_pairs(const float* __restrict__ s,
                                                          float* __restrict__ d) {
  size_t idx = (size_t)blockIdx.x * 256 + threadIdx.x;
  int c = (int)(idx & 127);
  int j = (int)((idx >> 7) & 255);
  int i = (int)(idx >> 15);
  d[idx] = s[(((size_t)j * 256 + i) << 7) + c];
}
__global__ void __launch_bounds__(256) pt_add_transpose(float* __restrict__ z,
                                                        const float* __restrict__ dlt) {
  size_t idx = (size_t)blockIdx.x * 256 + threadIdx.x;
  int c = (int)(idx & 127);
  int j = (int)((idx >> 7) & 255);
  int i = (int)(idx >> 15);
  z[idx] += dlt[(((size_t)j * 256 + i) << 7) + c];
}
__global__ void __launch_bounds__(256) pt_transpose_mask(const float* __restrict__ m,
                                                         float* __restrict__ mt) {
  int idx = blockIdx.x * 256 + threadIdx.x;
  int i = idx >> 8, j = idx & 255;
  mt[idx] = m[j * 256 + i];
}

// ---------------------------------------------------------------------------
// Host-side orchestration
// ---------------------------------------------------------------------------
extern "C" void kernel_launch(void* const* d_in, const int* in_sizes, int n_in,
                              void* d_out, int out_size, void* d_ws, size_t ws_size,
                              hipStream_t stream)
{
  (void)in_sizes; (void)n_in; (void)out_size; (void)ws_size;

  auto in = [&](int i) -> const float* { return (const float*)d_in[i]; };
  const float* p     = in(0);
  const float* maskp = in(1);
  float* z = (float*)d_out;

  const int    M  = 65536;
  const size_t MZ = (size_t)M * 128;

  float* wsf  = (float*)d_ws;
  float* zln  = wsf;
  float* b1   = wsf + 1 * MZ;
  float* b2   = wsf + 2 * MZ;
  float* b3   = wsf + 3 * MZ;
  float* b4   = wsf + 4 * MZ;
  float* b5   = wsf + 5 * MZ;
  float* hbuf = b2;                        // [M,512] aliases b2..b5 (free then)
  float* tb2  = wsf + 6 * MZ;              // [4][65536]
  float* maskT = tb2 + 4 * 65536;          // [65536]
  bf16_t* wpk  = (bf16_t*)(maskT + 65536); // packed-weight scratch (128 KB)

  dim3 blk(256);
  unsigned gMZ = (unsigned)(MZ / 256);
  dim3 gG(M / 32);
  dim3 gT(256, 8, 4);

  auto LN = [&](const float* s, float* d, const float* g, const float* b) {
    pt_layernorm<<<dim3(M / 8), blk, 0, stream>>>(s, d, g, b, M);
  };
  auto pack = [&](const float* W, int K, int NC) {
    pt_pack_weight<<<dim3((K * NC) / 256), blk, 0, stream>>>(W, wpk, K, NC);
  };

  // z = p
  pt_copy<<<dim3(gMZ), blk, 0, stream>>>(p, z, MZ);

  // ---- Triangle multiplicative update (param base pb; mode 0=out, 1=in) ----
  auto tri_mul_phase = [&](int pb, int mode) {
    LN(z, zln, in(pb), in(pb + 1));
    pack(in(pb + 4), 128, 128);
    pt_gemm<128, 128, ACT_SIG, true, false, false, false>
        <<<gG, blk, 0, stream>>>(zln, wpk, in(pb + 5), nullptr, nullptr, b1, M);
    pack(in(pb + 2), 128, 128);
    pt_gemm<128, 128, ACT_NONE, true, true, true, false>
        <<<gG, blk, 0, stream>>>(zln, wpk, in(pb + 3), b1, maskp, b2, M);
    pack(in(pb + 8), 128, 128);
    pt_gemm<128, 128, ACT_SIG, true, false, false, false>
        <<<gG, blk, 0, stream>>>(zln, wpk, in(pb + 9), nullptr, nullptr, b1, M);
    pack(in(pb + 6), 128, 128);
    pt_gemm<128, 128, ACT_NONE, true, true, true, false>
        <<<gG, blk, 0, stream>>>(zln, wpk, in(pb + 7), b1, maskp, b3, M);
    pt_transpose_cf<<<gT, blk, 0, stream>>>(b2, b4, mode);              // at
    pt_transpose_cf<<<gT, blk, 0, stream>>>(b3, b5, mode);              // bt
    pt_tri_gemm<<<dim3(128, 16, 2), blk, 0, stream>>>(b4, b5, b2);      // x (cf)
    pt_transpose_fc<<<gT, blk, 0, stream>>>(b2, b3);                    // x (cl)
    LN(b3, b2, in(pb + 10), in(pb + 11));                               // xln -> b2
    pack(in(pb + 12), 128, 128);
    pt_gemm<128, 128, ACT_SIG, true, false, false, false>
        <<<gG, blk, 0, stream>>>(zln, wpk, in(pb + 13), nullptr, nullptr, b1, M);
    pack(in(pb + 14), 128, 128);
    pt_gemm<128, 128, ACT_NONE, true, true, false, true>
        <<<gG, blk, 0, stream>>>(b2, wpk, in(pb + 15), b1, nullptr, z, M);
  };
  tri_mul_phase(2, 0);    // mul_out (outgoing)
  tri_mul_phase(18, 1);   // mul_in  (incoming)

  // ---- Triangle attention (param base pb) ----
  auto att_phase = [&](int pb, const float* zin, const float* msk,
                       float* obuf, float* dst, bool accum) {
    LN(zin, zln, in(pb), in(pb + 1));
    pack(in(pb + 2), 128, 128);
    pt_gemm<128, 128, ACT_NONE, false, false, false, false>
        <<<gG, blk, 0, stream>>>(zln, wpk, nullptr, nullptr, nullptr, b3, M); // q
    pack(in(pb + 3), 128, 128);
    pt_gemm<128, 128, ACT_NONE, false, false, false, false>
        <<<gG, blk, 0, stream>>>(zln, wpk, nullptr, nullptr, nullptr, b4, M); // k
    pack(in(pb + 4), 128, 128);
    pt_gemm<128, 128, ACT_NONE, false, false, false, false>
        <<<gG, blk, 0, stream>>>(zln, wpk, nullptr, nullptr, nullptr, b5, M); // v
    pack(in(pb + 6), 128, 128);
    pt_gemm<128, 128, ACT_SIG, true, false, false, false>
        <<<gG, blk, 0, stream>>>(zln, wpk, in(pb + 7), nullptr, nullptr, b1, M); // gate
    pt_tribias<<<dim3(M / 8), blk, 0, stream>>>(zln, in(pb + 5), tb2);
    pt_attention<<<dim3(256, 4), blk, 0, stream>>>(b3, b4, b5, tb2, msk, obuf);
    pt_mul<<<dim3(gMZ), blk, 0, stream>>>(obuf, b1, MZ);                // o *= gate
    pack(in(pb + 8), 128, 128);
    if (accum)
      pt_gemm<128, 128, ACT_NONE, true, false, false, true>
          <<<gG, blk, 0, stream>>>(obuf, wpk, in(pb + 9), nullptr, nullptr, dst, M);
    else
      pt_gemm<128, 128, ACT_NONE, true, false, false, false>
          <<<gG, blk, 0, stream>>>(obuf, wpk, in(pb + 9), nullptr, nullptr, dst, M);
  };

  // att_start: rows attend, delta added directly into z
  att_phase(34, z, maskp, b2, z, true);

  // att_end: columns attend (transposed space, add transposed delta)
  pt_transpose_pairs<<<dim3(gMZ), blk, 0, stream>>>(z, b2);             // zt
  pt_transpose_mask<<<dim3(256), blk, 0, stream>>>(maskp, maskT);
  att_phase(44, b2, maskT, b2, b3, false);                              // delta -> b3
  pt_add_transpose<<<dim3(gMZ), blk, 0, stream>>>(z, b3);               // z += T(delta)

  // ---- Transition MLP ----
  LN(z, zln, in(54), in(55));
  pack(in(56), 128, 512);
  pt_gemm<128, 512, ACT_RELU, true, false, false, false>
      <<<gG, blk, 0, stream>>>(zln, wpk, in(57), nullptr, nullptr, hbuf, M);
  pack(in(58), 512, 128);
  pt_gemm<512, 128, ACT_NONE, true, false, true, true>
      <<<gG, blk, 0, stream>>>(hbuf, wpk, in(59), nullptr, maskp, z, M);

  // final z *= mask
  pt_mask_rows<<<dim3(gMZ), blk, 0, stream>>>(z, maskp, MZ);
}
